// KpDetModule_47390669144784
// MI455X (gfx1250) — matrix operand order repaired and checked by reading, not compiled, not run on hardware
//
#include <hip/hip_runtime.h>
#include <math.h>

typedef __attribute__((ext_vector_type(16))) _Float16 v16h;
typedef __attribute__((ext_vector_type(8)))  _Float16 v8h;
typedef __attribute__((ext_vector_type(4)))  _Float16 v4h;
typedef __attribute__((ext_vector_type(8)))  float    v8f;
typedef __attribute__((ext_vector_type(4)))  float    v4f;

#define HWPIX 4096
#define FAM_ELEMS 5439488   // 256 * 21248 f16 elements per weight family

// ---------------------------------------------------------------------------
// Kernel 1a: deform_offset[166][4096] = w_do(166x34) @ kpt_prev(34x4096) + b_do
// ---------------------------------------------------------------------------
__global__ __launch_bounds__(256) void k_doff(const float* __restrict__ kp,
                                              const float* __restrict__ w_do,
                                              const float* __restrict__ b_do,
                                              float* __restrict__ doff) {
  int t = blockIdx.x * 256 + threadIdx.x;
  if (t >= 166 * HWPIX) return;
  int o = t >> 12;
  int pix = t & 4095;
  float s = b_do[o];
  #pragma unroll 2
  for (int c = 0; c < 34; ++c) s += w_do[o * 34 + c] * kp[c * HWPIX + pix];
  doff[t] = s;
}

// ---------------------------------------------------------------------------
// Kernel 1b: weight transpose+convert: dst[o][r] (f16), r = p*256 + c,
//            from src[o][c][p] (fp32, shape 256 x 256 x Ks)
// ---------------------------------------------------------------------------
__global__ __launch_bounds__(256) void k_wconv(const float* __restrict__ src,
                                               _Float16* __restrict__ dst,
                                               int Ks) {
  int Kred = 256 * Ks;
  int t = blockIdx.x * 256 + threadIdx.x;
  if (t >= 256 * Kred) return;
  int o = t / Kred;
  int r = t - o * Kred;
  int p = r >> 8;
  int c = r & 255;
  dst[t] = (_Float16)src[(o * 256 + c) * Ks + p];
}

// ---------------------------------------------------------------------------
// Kernel 1c: feature transpose to channel-last: dst[pix][c] = src[c][pix]
// ---------------------------------------------------------------------------
__global__ __launch_bounds__(256) void k_trans(const float* __restrict__ src,
                                               float* __restrict__ dst) {
  int t = blockIdx.x * 256 + threadIdx.x;  // < 4096*256
  int pix = t >> 8;
  int c = t & 255;
  dst[t] = src[c * HWPIX + pix];
}

// ---------------------------------------------------------------------------
// Kernel 2: fused deformable-conv (implicit GEMM via WMMA) + 1x1 heads + bbox
//   blockIdx.y == 0 : cls path (family Wc -> w_cls)
//   blockIdx.y == 1 : pts path (families Wk -> w_kpt, Wb -> w_bbox)
//   blockIdx.x : 32-pixel tile (4096/32 = 128 tiles)
// ---------------------------------------------------------------------------
__global__ __launch_bounds__(256) void k_fused(
    const float* __restrict__ ft_cls, const float* __restrict__ ft_pts,
    const float* __restrict__ doff,
    const _Float16* __restrict__ Wc, const _Float16* __restrict__ Wk,
    const _Float16* __restrict__ Wb,
    const float* __restrict__ w_cls, const float* __restrict__ b_cls,
    const float* __restrict__ w_kpt, const float* __restrict__ b_kpt,
    const float* __restrict__ w_bbox, const float* __restrict__ b_bbox,
    const float* __restrict__ kpt_prev, float* __restrict__ out) {
  __shared__ float s_wy[1568];
  __shared__ float s_wx[1568];
  __shared__ short s_y0[1568];
  __shared__ short s_x0[1568];
  __shared__ __align__(32) _Float16 sB[32 * 32];  // [n32][kc] sampled tile
  __shared__ float sD[256 * 32];                  // relu'd dfm tile [ch][n32]
  __shared__ float sKpt[34 * 32];
  __shared__ float sBB[4 * 32];

  const int mode = blockIdx.y;
  const float* __restrict__ ft = mode ? ft_pts : ft_cls;  // channel-last [pix][c]
  const int tile = blockIdx.x;
  const int t = threadIdx.x;
  const int lane = t & 31;
  const int wave = t >> 5;
  const int hi = lane >> 4;
  const int ln = lane & 15;

  float accR[5] = {0.f, 0.f, 0.f, 0.f, 0.f};  // persistent 1x1-conv accumulators

  const int Kp_[3] = {9, 25, 49};
  const int Kred_[3] = {2304, 6400, 12544};
  const int segoff_[3] = {0, 589824, 2228224};
  const int segb_[3] = {0, 18, 68};

  for (int seg = 0; seg < 3; ++seg) {
    const int Kp = Kp_[seg];
    const int Kred = Kred_[seg];
    const _Float16* __restrict__ WA0 = (mode ? Wk : Wc) + segoff_[seg];
    const _Float16* __restrict__ WA1 = Wb + segoff_[seg];

    // ---- bilinear coordinate tables: Kp points x 32 pixels
    for (int e = t; e < Kp * 32; e += 256) {
      int p = e >> 5, n = e & 31;
      int pix = tile * 32 + n;
      int i = pix >> 6, j = pix & 63;
      float dy = doff[(size_t)(segb_[seg] + 2 * p) * HWPIX + pix];
      float dx = doff[(size_t)(segb_[seg] + 2 * p + 1) * HWPIX + pix];
      float y = (float)i + dy;
      float x = (float)j + dx;
      float y0 = floorf(y), x0 = floorf(x);
      s_wy[e] = y - y0;
      s_wx[e] = x - x0;
      s_y0[e] = (short)(int)y0;
      s_x0[e] = (short)(int)x0;
    }
    __syncthreads();

    v8f a000 = {}, a001 = {}, a010 = {}, a011 = {};
    v8f a100 = {}, a101 = {}, a110 = {}, a111 = {};
    const int nch = Kred >> 5;
    for (int chk = 0; chk < nch; ++chk) {
      const int rb = chk << 5;
      const int p = rb >> 8;
      const int cb = rb & 255;

      // ---- bilinear sampling: 32 px x 32 ch; each thread: 1 px, 4 channels
      {
        const int n32 = t & 31;           // pixel within tile
        const int co = (t >> 5) * 4;      // channel quad (wave-based)
        const int e = p * 32 + n32;
        const int c = cb + co;
        int y0 = s_y0[e], x0 = s_x0[e];
        float wy = s_wy[e], wx = s_wx[e];
        bool yi0 = (y0 >= 0) & (y0 < 64);
        bool yi1 = (y0 >= -1) & (y0 < 63);
        bool xi0 = (x0 >= 0) & (x0 < 64);
        bool xi1 = (x0 >= -1) & (x0 < 63);
        const float* __restrict__ fp = ft + c;
        size_t i00 = (size_t)(y0 * 64 + x0) * 256;
        v4f zero = (v4f)0.f;
        v4f v00 = (yi0 & xi0) ? *(const v4f*)(fp + i00) : zero;
        v4f v01 = (yi0 & xi1) ? *(const v4f*)(fp + i00 + 256) : zero;
        v4f v10 = (yi1 & xi0) ? *(const v4f*)(fp + i00 + 64 * 256) : zero;
        v4f v11 = (yi1 & xi1) ? *(const v4f*)(fp + i00 + 65 * 256) : zero;
        v4f wxv = (v4f)wx, wyv = (v4f)wy;
        v4f vx0 = v00 + wxv * (v01 - v00);
        v4f vx1 = v10 + wxv * (v11 - v10);
        v4f sv = vx0 + wyv * (vx1 - vx0);
        v4h h;
        h.x = (_Float16)sv.x; h.y = (_Float16)sv.y;
        h.z = (_Float16)sv.z; h.w = (_Float16)sv.w;
        *(v4h*)&sB[n32 * 32 + co] = h;
      }
      __syncthreads();

      // ---- WMMA: two B sub-tiles (pixels 0-15, 16-31), A reused across both
      v16h bm0, bm1;
      {
        v8h b0l = *(const v8h*)&sB[(ln)*32 + hi * 8];
        v8h b0h = *(const v8h*)&sB[(ln)*32 + hi * 8 + 16];
        v8h b1l = *(const v8h*)&sB[(16 + ln) * 32 + hi * 8];
        v8h b1h = *(const v8h*)&sB[(16 + ln) * 32 + hi * 8 + 16];
        #pragma unroll
        for (int q = 0; q < 8; ++q) { bm0[q] = b0l[q]; bm0[8 + q] = b0h[q]; bm1[q] = b1l[q]; bm1[8 + q] = b1h[q]; }
      }
      {
        const int m0 = wave * 32 + ln;
        const size_t ba0 = (size_t)m0 * Kred + (size_t)(rb + hi * 8);
        const size_t ba1 = (size_t)(m0 + 16) * Kred + (size_t)(rb + hi * 8);
        v8h al = *(const v8h*)(WA0 + ba0);
        v8h ah = *(const v8h*)(WA0 + ba0 + 16);
        v16h am;
        #pragma unroll
        for (int q = 0; q < 8; ++q) { am[q] = al[q]; am[8 + q] = ah[q]; }
        a000 = __builtin_amdgcn_wmma_f32_16x16x32_f16(false, am, false, bm0,
                                                      (short)0, a000, false, false);
        a001 = __builtin_amdgcn_wmma_f32_16x16x32_f16(false, am, false, bm1,
                                                      (short)0, a001, false, false);
        v8h bl = *(const v8h*)(WA0 + ba1);
        v8h bh = *(const v8h*)(WA0 + ba1 + 16);
        v16h am2;
        #pragma unroll
        for (int q = 0; q < 8; ++q) { am2[q] = bl[q]; am2[8 + q] = bh[q]; }
        a010 = __builtin_amdgcn_wmma_f32_16x16x32_f16(false, am2, false, bm0,
                                                      (short)0, a010, false, false);
        a011 = __builtin_amdgcn_wmma_f32_16x16x32_f16(false, am2, false, bm1,
                                                      (short)0, a011, false, false);
        if (mode) {
          v8h cl = *(const v8h*)(WA1 + ba0);
          v8h chh = *(const v8h*)(WA1 + ba0 + 16);
          v16h am3;
          #pragma unroll
          for (int q = 0; q < 8; ++q) { am3[q] = cl[q]; am3[8 + q] = chh[q]; }
          a100 = __builtin_amdgcn_wmma_f32_16x16x32_f16(false, am3, false, bm0,
                                                        (short)0, a100, false, false);
          a101 = __builtin_amdgcn_wmma_f32_16x16x32_f16(false, am3, false, bm1,
                                                        (short)0, a101, false, false);
          v8h dl = *(const v8h*)(WA1 + ba1);
          v8h dh = *(const v8h*)(WA1 + ba1 + 16);
          v16h am4;
          #pragma unroll
          for (int q = 0; q < 8; ++q) { am4[q] = dl[q]; am4[8 + q] = dh[q]; }
          a110 = __builtin_amdgcn_wmma_f32_16x16x32_f16(false, am4, false, bm0,
                                                        (short)0, a110, false, false);
          a111 = __builtin_amdgcn_wmma_f32_16x16x32_f16(false, am4, false, bm1,
                                                        (short)0, a111, false, false);
        }
      }
      __syncthreads();
    }

    // ---- epilogue family 0: relu -> LDS tile (m = q + 8*hi, col = nt*16+ln)
    #pragma unroll
    for (int q = 0; q < 8; ++q) {
      int chn = wave * 32 + q + 8 * hi;
      sD[chn * 32 + ln] = fmaxf(a000[q], 0.f);
      sD[chn * 32 + 16 + ln] = fmaxf(a001[q], 0.f);
      sD[(chn + 16) * 32 + ln] = fmaxf(a010[q], 0.f);
      sD[(chn + 16) * 32 + 16 + ln] = fmaxf(a011[q], 0.f);
    }
    __syncthreads();
    if (mode == 0) {
      #pragma unroll
      for (int s2 = 0; s2 < 2; ++s2) {
        int q = t + s2 * 256;
        if (q < 416) {  // 13 cls outputs x 32 px
          int o = q >> 5, n = q & 31;
          const float* __restrict__ wr = w_cls + o * 768 + seg * 256;
          float s = 0.f;
          #pragma unroll 8
          for (int c = 0; c < 256; ++c) s += sD[c * 32 + n] * wr[c];
          accR[s2] += s;
        }
      }
    } else {
      #pragma unroll
      for (int s2 = 0; s2 < 5; ++s2) {
        int q = t + s2 * 256;
        if (q < 1088) {  // 34 kpt outputs x 32 px
          int o = q >> 5, n = q & 31;
          const float* __restrict__ wr = w_kpt + o * 768 + seg * 256;
          float s = 0.f;
          #pragma unroll 8
          for (int c = 0; c < 256; ++c) s += sD[c * 32 + n] * wr[c];
          accR[s2] += s;
        }
      }
    }
    __syncthreads();
    if (mode) {
      // ---- epilogue family 1 (bbox weights)
      #pragma unroll
      for (int q = 0; q < 8; ++q) {
        int chn = wave * 32 + q + 8 * hi;
        sD[chn * 32 + ln] = fmaxf(a100[q], 0.f);
        sD[chn * 32 + 16 + ln] = fmaxf(a101[q], 0.f);
        sD[(chn + 16) * 32 + ln] = fmaxf(a110[q], 0.f);
        sD[(chn + 16) * 32 + 16 + ln] = fmaxf(a111[q], 0.f);
      }
      __syncthreads();
      {
        int q = t + 1024;  // only slot 4 covers the bbox pair range
        if (q >= 1088 && q < 1216) {  // 4 bbox params x 32 px
          int qq = q - 1088;
          int o = qq >> 5, n = qq & 31;
          const float* __restrict__ wr = w_bbox + o * 768 + seg * 256;
          float s = 0.f;
          #pragma unroll 8
          for (int c = 0; c < 256; ++c) s += sD[c * 32 + n] * wr[c];
          accR[4] += s;
        }
      }
      __syncthreads();
    }
  }

  // ---- final outputs
  // d_out layout: cls [0, 53248), bbox [53248, 69632), kpt [69632, 208896)
  if (mode == 0) {
    #pragma unroll
    for (int s2 = 0; s2 < 2; ++s2) {
      int q = t + s2 * 256;
      if (q < 416) {
        int o = q >> 5, n = q & 31;
        out[(size_t)o * HWPIX + tile * 32 + n] = accR[s2] + b_cls[o];
      }
    }
  } else {
    #pragma unroll
    for (int s2 = 0; s2 < 5; ++s2) {
      int q = t + s2 * 256;
      if (q < 1088) {
        int o = q >> 5, n = q & 31;
        int pix = tile * 32 + n;
        float v = accR[s2] + b_kpt[o] + kpt_prev[(size_t)o * HWPIX + pix];
        out[69632 + (size_t)o * HWPIX + pix] = v;
        sKpt[o * 32 + n] = v;
      } else if (q < 1216) {
        int qq = q - 1088;
        int o = qq >> 5, n = qq & 31;
        sBB[o * 32 + n] = accR[s2] + b_bbox[o];
      }
    }
    __syncthreads();
    if (t < 32) {
      int n = t;
      int pix = tile * 32 + n;
      float left = 1e30f, right = -1e30f, top = 1e30f, bot = -1e30f;
      #pragma unroll
      for (int p = 0; p < 17; ++p) {
        float py = sKpt[(2 * p) * 32 + n];
        float qx = sKpt[(2 * p + 1) * 32 + n];
        left = fminf(left, qx);
        right = fmaxf(right, qx);
        top = fminf(top, py);
        bot = fmaxf(bot, py);
      }
      float cx = 0.5f * (left + right);
      float cy = 0.5f * (top + bot);
      float hw = (cx - left) * expf(sBB[0 * 32 + n]);
      float hh = (cy - top) * expf(sBB[1 * 32 + n]);
      cx += sBB[2 * 32 + n];
      cy += sBB[3 * 32 + n];
      out[53248 + 0 * HWPIX + pix] = cx - hw;
      out[53248 + 1 * HWPIX + pix] = cy - hh;
      out[53248 + 2 * HWPIX + pix] = cx + hw;
      out[53248 + 3 * HWPIX + pix] = cy + hh;
    }
  }
}

// ---------------------------------------------------------------------------
extern "C" void kernel_launch(void* const* d_in, const int* in_sizes, int n_in,
                              void* d_out, int out_size, void* d_ws, size_t ws_size,
                              hipStream_t stream) {
  const float* cls_feat = (const float*)d_in[0];
  const float* pts_feat = (const float*)d_in[1];
  const float* kpt_prev = (const float*)d_in[2];
  const float* w_do = (const float*)d_in[3];
  const float* b_do = (const float*)d_in[4];
  // weights d_in[5..13]: w_c3,w_c5,w_c7, w_k3,w_k5,w_k7, w_b3,w_b5,w_b7
  const float* w_cls = (const float*)d_in[14];
  const float* b_cls = (const float*)d_in[15];
  const float* w_kpt = (const float*)d_in[16];
  const float* b_kpt = (const float*)d_in[17];
  const float* w_bbox = (const float*)d_in[18];
  const float* b_bbox = (const float*)d_in[19];
  float* out = (float*)d_out;

  // workspace layout
  float* doff = (float*)d_ws;                                   // 166*4096 f32
  _Float16* Wf = (_Float16*)((char*)d_ws + 2719744);            // 3 fam * FAM_ELEMS f16
  float* ft_cls = (float*)((char*)d_ws + 35356672);             // 4096*256 f32
  float* ft_pts = (float*)((char*)d_ws + 35356672 + 4194304);   // 4096*256 f32

  // 1a: deform offsets
  k_doff<<<(166 * HWPIX + 255) / 256, 256, 0, stream>>>(kpt_prev, w_do, b_do, doff);

  // 1b: weight transpose/convert to f16 [o][p*256+c]
  const int Ks_[3] = {9, 25, 49};
  const size_t segoff_[3] = {0, 589824, 2228224};
  for (int f = 0; f < 3; ++f) {
    for (int s = 0; s < 3; ++s) {
      const float* src = (const float*)d_in[5 + f * 3 + s];
      _Float16* dst = Wf + (size_t)f * FAM_ELEMS + segoff_[s];
      int nel = 256 * 256 * Ks_[s];
      k_wconv<<<(nel + 255) / 256, 256, 0, stream>>>(src, dst, Ks_[s]);
    }
  }

  // 1c: channel-last feature transposes
  k_trans<<<(HWPIX * 256) / 256, 256, 0, stream>>>(cls_feat, ft_cls);
  k_trans<<<(HWPIX * 256) / 256, 256, 0, stream>>>(pts_feat, ft_pts);

  // 2: fused deformable conv + heads
  k_fused<<<dim3(128, 2), 256, 0, stream>>>(
      ft_cls, ft_pts, doff,
      Wf, Wf + FAM_ELEMS, Wf + 2 * (size_t)FAM_ELEMS,
      w_cls, b_cls, w_kpt, b_kpt, w_bbox, b_bbox, kpt_prev, out);
}
